// SAMMEncoderV8_80616536145994
// MI455X (gfx1250) — matrix operand order repaired and checked by reading, not compiled, not run on hardware
//
#include <hip/hip_runtime.h>
#include <hip/hip_bf16.h>
#include <math.h>

// ---------------------------------------------------------------------------
// CDNA5 (gfx1250) transformer encoder pipeline, all-bf16 GEMM operands.
//  - v_wmma_f32_16x16x32_bf16 (wave32, f32 accumulate)
//  - large GEMMs: block-cooperative A-tile staging in LDS via
//    global_load_async_to_lds_b128 + s_wait_asynccnt (double buffered)
//  - small GEMM (attn@V, N=64): direct-global b128 loads
// ---------------------------------------------------------------------------

typedef __attribute__((ext_vector_type(16))) __bf16 v16bf;
typedef __attribute__((ext_vector_type(8)))  float  v8f;

#define DEVINL __device__ __forceinline__

DEVINL __bf16 f2bf(float f) {
  unsigned u = __builtin_bit_cast(unsigned, f);
  unsigned r = u + 0x7FFFu + ((u >> 16) & 1u);   // round-to-nearest-even
  unsigned short h = (unsigned short)(r >> 16);
  return __builtin_bit_cast(__bf16, h);
}

DEVINL float gelu_exact(float x) {
  return 0.5f * x * (1.0f + erff(x * 0.70710678118654752f));
}

DEVINL float waveSum(float v) {
  #pragma unroll
  for (int m = 16; m; m >>= 1) v += __shfl_xor(v, m, 32);
  return v;
}
DEVINL float waveMax(float v) {
  #pragma unroll
  for (int m = 16; m; m >>= 1) v = fmaxf(v, __shfl_xor(v, m, 32));
  return v;
}

struct bfpack16 { uint4 lo, hi; };              // 16 x bf16 = 32 bytes
DEVINL v16bf load16(const __bf16* p0, const __bf16* p1) {
  bfpack16 t;
  t.lo = *(const uint4*)p0;
  t.hi = *(const uint4*)p1;
  return __builtin_bit_cast(v16bf, t);
}

DEVINL void storeC(float* p, float v)  { *p = v; }
DEVINL void storeC(__bf16* p, float v) { *p = f2bf(v); }

template<typename T> DEVINL T toOut(float v);
template<> DEVINL float  toOut<float >(float v) { return v; }
template<> DEVINL __bf16 toOut<__bf16>(float v) { return f2bf(v); }

// Shared epilogue: scale, bias, residual, GELU, store (fp32 or bf16).
template<bool GELU, typename CT>
DEVINL void gemm_epilogue(CT* __restrict__ Cb, const float* __restrict__ Rb,
                          const float* __restrict__ bias, float sc,
                          int tm, int tn, int half, int r, long sCm,
                          const v8f& c00, const v8f& c01,
                          const v8f& c10, const v8f& c11)
{
  const int n0 = tn + r, n1 = n0 + 16;
  const float bi0 = bias ? bias[n0] : 0.0f;
  const float bi1 = bias ? bias[n1] : 0.0f;
  #pragma unroll
  for (int j = 0; j < 8; ++j) {
    const long m0 = tm + half * 8 + j;           // C layout: half0 M=j, half1 M=8+j
    const long m1 = m0 + 16;
    float v00 = c00[j] * sc + bi0;
    float v01 = c01[j] * sc + bi1;
    float v10 = c10[j] * sc + bi0;
    float v11 = c11[j] * sc + bi1;
    if (Rb) {
      v00 += Rb[m0 * sCm + n0]; v01 += Rb[m0 * sCm + n1];
      v10 += Rb[m1 * sCm + n0]; v11 += Rb[m1 * sCm + n1];
    }
    if (GELU) {
      v00 = gelu_exact(v00); v01 = gelu_exact(v01);
      v10 = gelu_exact(v10); v11 = gelu_exact(v11);
    }
    storeC(&Cb[m0 * sCm + n0], v00);  storeC(&Cb[m0 * sCm + n1], v01);
    storeC(&Cb[m1 * sCm + n0], v10);  storeC(&Cb[m1 * sCm + n1], v11);
  }
}

// ---------------------------------------------------------------------------
// LDS-staged batched WMMA GEMM (requires N%256==0, K%64==0).
// Block = 8 waves sharing A rows [tm, tm+32); wave w owns N cols
// [tnBase + 32w, +32). Per 64-deep K-slab every wave issues ONE
// global_load_async_to_lds_b128 (32 lanes x 16B = 512B; 8 waves = full
// 32x64 bf16 tile), waits ASYNCcnt, barrier, then computes two K=32 WMMA
// steps with A fragments from LDS. Double buffered.
// ---------------------------------------------------------------------------
template<bool BCONTIG, bool GELU, typename CT>
__global__ __launch_bounds__(256)
void wmma_gemm_lds(const __bf16* __restrict__ A, const __bf16* __restrict__ B,
                   CT* __restrict__ C,
                   const float* __restrict__ bias,
                   const float* __restrict__ resid,
                   const float* __restrict__ scalePtr,
                   int M, int N, int K,
                   long sAm, long sAb,
                   long sBk, long sBn, long sBb,
                   long sCm, long sCb, float scale)
{
  __shared__ __bf16 tileA[2][32 * 64];           // 2 x 4KB

  const int lane = threadIdx.x & 31;
  const int w    = threadIdx.x >> 5;
  const int nGrp = N >> 8;                       // N / 256
  const int tm   = ((int)blockIdx.x / nGrp) << 5;
  const int tn   = (((int)blockIdx.x % nGrp) << 8) + (w << 5);

  const long batch = blockIdx.y;
  const __bf16* Ab = A + batch * sAb;
  const __bf16* Bb = B + batch * sBb;
  CT*           Cb = C + batch * sCb;
  const float*  Rb = resid ? resid + batch * sCb : nullptr;

  const int half = lane >> 4, r = lane & 15;
  const __bf16* b0p = Bb + (long)(tn + r) * sBn;
  const __bf16* b1p = b0p + 16 * sBn;

  // This thread's async-copy assignment: chunk c = w*32+lane of 256 chunks,
  // 16B each: row = c/8 (0..31), kchunk = c%8 (8 bf16 = 16B).
  const int c16  = (w << 5) | lane;
  const int arow = c16 >> 3, akc = c16 & 7;
  const __bf16* asrc = Ab + (long)(tm + arow) * sAm + akc * 8;
  const unsigned adst0 = (unsigned)(size_t)(&tileA[0][arow * 64 + akc * 8]);
  const unsigned adst1 = (unsigned)(size_t)(&tileA[1][arow * 64 + akc * 8]);

  v8f c00 = {}, c01 = {}, c10 = {}, c11 = {};
  const int nSlab = K >> 6;

  // prologue: slab 0 -> buf 0
  asm volatile("global_load_async_to_lds_b128 %0, %1, off"
               :: "v"(adst0), "v"(asrc) : "memory");

  for (int s = 0; s < nSlab; ++s) {
    const int buf = s & 1;
    asm volatile("s_wait_asynccnt 0" ::: "memory");   // own async done
    __syncthreads();                                  // everyone's async done
    if (s + 1 < nSlab) {                              // overlap next slab copy
      const __bf16* g = asrc + (long)(s + 1) * 64;
      asm volatile("global_load_async_to_lds_b128 %0, %1, off"
                   :: "v"(buf ? adst0 : adst1), "v"(g) : "memory");
    }
    #pragma unroll
    for (int kk = 0; kk < 64; kk += 32) {
      const int ka = kk + half * 16;
      const __bf16* la0 = &tileA[buf][r * 64 + ka];
      const __bf16* la1 = &tileA[buf][(r + 16) * 64 + ka];
      v16bf a0 = load16(la0, la0 + 8);                // ds_load_b128 pairs
      v16bf a1 = load16(la1, la1 + 8);
      const long kb = (long)(s << 6) + kk + half * 16;
      v16bf b0, b1;
      if (BCONTIG) {
        b0 = load16(b0p + kb, b0p + kb + 8);
        b1 = load16(b1p + kb, b1p + kb + 8);
      } else {
        #pragma unroll
        for (int i = 0; i < 16; ++i) {
          const long kx = (kb + i) * sBk;
          b0[i] = b0p[kx];
          b1[i] = b1p[kx];
        }
      }
      c00 = __builtin_amdgcn_wmma_f32_16x16x32_bf16(false, a0, false, b0, (short)0, c00, false, false);
      c01 = __builtin_amdgcn_wmma_f32_16x16x32_bf16(false, a0, false, b1, (short)0, c01, false, false);
      c10 = __builtin_amdgcn_wmma_f32_16x16x32_bf16(false, a1, false, b0, (short)0, c10, false, false);
      c11 = __builtin_amdgcn_wmma_f32_16x16x32_bf16(false, a1, false, b1, (short)0, c11, false, false);
    }
    __syncthreads();                                  // buf consumed by all
  }

  const float sc = scale * (scalePtr ? scalePtr[0] : 1.0f);
  gemm_epilogue<GELU, CT>(Cb, Rb, bias, sc, tm, tn, half, r, sCm, c00, c01, c10, c11);
}

// ---------------------------------------------------------------------------
// Direct-global batched WMMA GEMM (fallback; used for attn@V, N=64).
// ---------------------------------------------------------------------------
template<bool BCONTIG, bool GELU, typename CT>
__global__ __launch_bounds__(256)
void wmma_gemm(const __bf16* __restrict__ A, const __bf16* __restrict__ B,
               CT* __restrict__ C,
               const float* __restrict__ bias,
               const float* __restrict__ resid,
               const float* __restrict__ scalePtr,
               int M, int N, int K,
               long sAm, long sAb,
               long sBk, long sBn, long sBb,
               long sCm, long sCb, float scale)
{
  const int lane = threadIdx.x & 31;
  const int wv   = threadIdx.x >> 5;
  const int tilesN = N >> 5;
  const int tilesPerBatch = (M >> 5) * tilesN;
  const int tile = blockIdx.x * (blockDim.x >> 5) + wv;
  if (tile >= tilesPerBatch) return;

  const long batch = blockIdx.y;
  const __bf16* Ab = A + batch * sAb;
  const __bf16* Bb = B + batch * sBb;
  CT*           Cb = C + batch * sCb;
  const float*  Rb = resid ? resid + batch * sCb : nullptr;

  const int tm = (tile / tilesN) << 5;
  const int tn = (tile % tilesN) << 5;
  const int half = lane >> 4, r = lane & 15;

  const __bf16* a0p = Ab + (long)(tm + r) * sAm;
  const __bf16* a1p = a0p + 16 * sAm;
  const __bf16* b0p = Bb + (long)(tn + r) * sBn;
  const __bf16* b1p = b0p + 16 * sBn;

  v8f c00 = {}, c01 = {}, c10 = {}, c11 = {};

  for (int k0 = 0; k0 < K; k0 += 32) {
    if (k0 + 32 < K) {
      __builtin_prefetch(a0p + k0 + 32, 0, 0);
      __builtin_prefetch(b0p + (long)(k0 + 32) * sBk, 0, 0);
    }
    const int ka = k0 + half * 16;
    v16bf a0 = load16(a0p + ka, a0p + ka + 8);
    v16bf a1 = load16(a1p + ka, a1p + ka + 8);
    const long kb = k0 + half * 16;
    v16bf b0, b1;
    if (BCONTIG) {
      b0 = load16(b0p + kb, b0p + kb + 8);
      b1 = load16(b1p + kb, b1p + kb + 8);
    } else {
      #pragma unroll
      for (int i = 0; i < 16; ++i) {
        const long kx = (kb + i) * sBk;
        b0[i] = b0p[kx];
        b1[i] = b1p[kx];
      }
    }
    c00 = __builtin_amdgcn_wmma_f32_16x16x32_bf16(false, a0, false, b0, (short)0, c00, false, false);
    c01 = __builtin_amdgcn_wmma_f32_16x16x32_bf16(false, a0, false, b1, (short)0, c01, false, false);
    c10 = __builtin_amdgcn_wmma_f32_16x16x32_bf16(false, a1, false, b0, (short)0, c10, false, false);
    c11 = __builtin_amdgcn_wmma_f32_16x16x32_bf16(false, a1, false, b1, (short)0, c11, false, false);
  }

  const float sc = scale * (scalePtr ? scalePtr[0] : 1.0f);
  gemm_epilogue<GELU, CT>(Cb, Rb, bias, sc, tm, tn, half, r, sCm, c00, c01, c10, c11);
}

// ---------------------------------------------------------------------------
// Row kernels (one wave per row, wave32 shfl reductions). f32 in, OutT out.
// ---------------------------------------------------------------------------
template<typename OutT>
__global__ void layernorm_rows(const float* __restrict__ x, const float* __restrict__ g,
                               const float* __restrict__ b, OutT* __restrict__ out,
                               int rows, int cols)
{
  const int wv = (int)((blockIdx.x * (long)blockDim.x + threadIdx.x) >> 5);
  if (wv >= rows) return;
  const int lane = threadIdx.x & 31;
  const float* xr = x + (long)wv * cols;
  float s = 0.f, ss = 0.f;
  for (int c = lane; c < cols; c += 32) { float v = xr[c]; s += v; ss += v * v; }
  s = waveSum(s); ss = waveSum(ss);
  const float mean = s / cols;
  const float var  = ss / cols - mean * mean;
  const float rstd = rsqrtf(var + 1e-5f);
  OutT* orow = out + (long)wv * cols;
  for (int c = lane; c < cols; c += 32)
    orow[c] = toOut<OutT>((xr[c] - mean) * rstd * g[c] + b[c]);
}

template<typename OutT>
__global__ void l2norm_rows(const float* __restrict__ x, OutT* __restrict__ out,
                            int rows, int cols)
{
  const int wv = (int)((blockIdx.x * (long)blockDim.x + threadIdx.x) >> 5);
  if (wv >= rows) return;
  const int lane = threadIdx.x & 31;
  const float* xr = x + (long)wv * cols;
  float ss = 0.f;
  for (int c = lane; c < cols; c += 32) { float v = xr[c]; ss += v * v; }
  ss = waveSum(ss);
  const float inv = 1.0f / fmaxf(sqrtf(ss), 1e-12f);
  OutT* orow = out + (long)wv * cols;
  for (int c = lane; c < cols; c += 32) orow[c] = toOut<OutT>(xr[c] * inv);
}

template<typename OutT>
__global__ void softmax_rows(const float* __restrict__ x, const float* __restrict__ g,
                             OutT* __restrict__ out, int rows, int cols,
                             float xs, float gs)
{
  const int wv = (int)((blockIdx.x * (long)blockDim.x + threadIdx.x) >> 5);
  if (wv >= rows) return;
  const int lane = threadIdx.x & 31;
  const float* xr = x + (long)wv * cols;
  const float* gr = g ? g + (long)wv * cols : nullptr;
  float mx = -3.4e38f;
  for (int c = lane; c < cols; c += 32) {
    float v = xr[c] * xs + (gr ? gr[c] * gs : 0.0f);
    mx = fmaxf(mx, v);
  }
  mx = waveMax(mx);
  float sum = 0.f;
  OutT* orow = out + (long)wv * cols;
  for (int c = lane; c < cols; c += 32)
    sum += __expf(xr[c] * xs + (gr ? gr[c] * gs : 0.0f) - mx);
  sum = waveSum(sum);
  const float inv = 1.0f / sum;
  for (int c = lane; c < cols; c += 32)
    orow[c] = toOut<OutT>(__expf(xr[c] * xs + (gr ? gr[c] * gs : 0.0f) - mx) * inv);
}

__global__ void cvt4_f32_bf16(const float* __restrict__ src, __bf16* __restrict__ dst, long n4)
{
  long i = blockIdx.x * (long)blockDim.x + threadIdx.x;
  if (i >= n4) return;
  float4 v = ((const float4*)src)[i];
  dst[4*i+0] = f2bf(v.x); dst[4*i+1] = f2bf(v.y);
  dst[4*i+2] = f2bf(v.z); dst[4*i+3] = f2bf(v.w);
}

__global__ void add_posenc(float* __restrict__ h, const float* __restrict__ pe,
                           long BT, int T, int D)
{
  long i = blockIdx.x * (long)blockDim.x + threadIdx.x;
  if (i >= BT * D) return;
  long row = i / D, c = i % D;
  h[i] += pe[(row % T) * (long)D + c];
}

__global__ void vq_scale_kernel(const int* __restrict__ epoch, const int* __restrict__ total,
                                float* __restrict__ out)
{
  float temp = fmaxf(0.5f, 1.0f - 0.5f * (float)epoch[0] / (float)total[0]);
  out[0] = 1.0f / temp;
}

__global__ void vq_finalize(const float* __restrict__ logits, const float* __restrict__ patterns,
                            float* __restrict__ emb, float* __restrict__ assign,
                            float* __restrict__ hard, int rows, int NP, int D)
{
  const int wv = (int)((blockIdx.x * (long)blockDim.x + threadIdx.x) >> 5);
  if (wv >= rows) return;
  const int lane = threadIdx.x & 31;
  const float* lr = logits + (long)wv * NP;
  float bm = -3.4e38f; int bi = 0x7fffffff;
  for (int c = lane; c < NP; c += 32) {
    float v = lr[c];
    if (v > bm || (v == bm && c < bi)) { bm = v; bi = c; }
  }
  #pragma unroll
  for (int m = 16; m; m >>= 1) {
    float om = __shfl_xor(bm, m, 32);
    int   oi = __shfl_xor(bi, m, 32);
    if (om > bm || (om == bm && oi < bi)) { bm = om; bi = oi; }
  }
  const float* pr = patterns + (long)bi * D;
  float* er = emb + (long)wv * D;
  for (int c = lane; c < D; c += 32) er[c] = pr[c];
  float* ar = assign + (long)wv * NP;
  for (int c = lane; c < NP; c += 32) ar[c] = (c == bi) ? 1.0f : 0.0f;
  if (lane == 0) hard[wv] = (float)bi;
}

// ---------------------------------------------------------------------------
// Host-side helpers
// ---------------------------------------------------------------------------
static void gemm(hipStream_t st, const __bf16* A, const __bf16* B, void* C, bool cbf16,
                 const float* bias, const float* resid, const float* scalePtr,
                 int M, int N, int K,
                 long sAm, long sAb, long sBk, long sBn, long sBb,
                 long sCm, long sCb, int batches, float scale, bool geluF)
{
  dim3 block(256, 1, 1);
  const bool bc = (sBk == 1);
  const bool staged = (N % 256 == 0) && (K % 64 == 0);
  if (staged) {
    dim3 grid((M / 32) * (N / 256), batches, 1);
#define GEMM_LDS(BC, GE, CT) \
    wmma_gemm_lds<BC, GE, CT><<<grid, block, 0, st>>>(A, B, (CT*)C, bias, resid, scalePtr, \
        M, N, K, sAm, sAb, sBk, sBn, sBb, sCm, sCb, scale)
    if (bc) {
      if (geluF) { if (cbf16) GEMM_LDS(true,  true,  __bf16); else GEMM_LDS(true,  true,  float); }
      else       { if (cbf16) GEMM_LDS(true,  false, __bf16); else GEMM_LDS(true,  false, float); }
    } else {
      if (geluF) { if (cbf16) GEMM_LDS(false, true,  __bf16); else GEMM_LDS(false, true,  float); }
      else       { if (cbf16) GEMM_LDS(false, false, __bf16); else GEMM_LDS(false, false, float); }
    }
#undef GEMM_LDS
  } else {
    dim3 grid(((M / 32) * (N / 32) + 7) / 8, batches, 1);
#define GEMM_GO(BC, GE, CT) \
    wmma_gemm<BC, GE, CT><<<grid, block, 0, st>>>(A, B, (CT*)C, bias, resid, scalePtr, \
        M, N, K, sAm, sAb, sBk, sBn, sBb, sCm, sCb, scale)
    if (bc) {
      if (geluF) { if (cbf16) GEMM_GO(true,  true,  __bf16); else GEMM_GO(true,  true,  float); }
      else       { if (cbf16) GEMM_GO(true,  false, __bf16); else GEMM_GO(true,  false, float); }
    } else {
      if (geluF) { if (cbf16) GEMM_GO(false, true,  __bf16); else GEMM_GO(false, true,  float); }
      else       { if (cbf16) GEMM_GO(false, false, __bf16); else GEMM_GO(false, false, float); }
    }
#undef GEMM_GO
  }
}

static inline int rowBlocks(int rows) { return (rows * 32 + 255) / 256; }

static void cvt(hipStream_t st, const float* src, __bf16* dst, long n) {
  long n4 = n / 4;
  cvt4_f32_bf16<<<(int)((n4 + 255) / 256), 256, 0, st>>>(src, dst, n4);
}

extern "C" void kernel_launch(void* const* d_in, const int* in_sizes, int n_in,
                              void* d_out, int out_size, void* d_ws, size_t ws_size,
                              hipStream_t stream)
{
  (void)in_sizes; (void)n_in; (void)out_size; (void)ws_size;
  const int B = 8, T = 1024, IN = 1024, D = 512, L = 4, NS = 1024, NP = 512, H = 8;
  const int BT = B * T;                  // 8192
  const long TQKV = (long)T * 3 * D;     // per-b qkv block (elements)

  const float* x        = (const float*)d_in[0];
  const float* gumbel   = (const float*)d_in[1];
  const float* sym_w    = (const float*)d_in[2];
  const float* sym_b    = (const float*)d_in[3];
  const float* codebook = (const float*)d_in[4];
  const float* pos_enc  = (const float*)d_in[5];
  const float* ain_w    = (const float*)d_in[6];
  const float* ain_b    = (const float*)d_in[7];
  const float* aout_w   = (const float*)d_in[8];
  const float* aout_b   = (const float*)d_in[9];
  const float* n1_g     = (const float*)d_in[10];
  const float* n1_b     = (const float*)d_in[11];
  const float* n2_g     = (const float*)d_in[12];
  const float* n2_b     = (const float*)d_in[13];
  const float* f_w1     = (const float*)d_in[14];
  const float* f_b1     = (const float*)d_in[15];
  const float* f_w2     = (const float*)d_in[16];
  const float* f_b2     = (const float*)d_in[17];
  const float* q_w      = (const float*)d_in[18];
  const float* q_b      = (const float*)d_in[19];
  const float* qln_g    = (const float*)d_in[20];
  const float* qln_b    = (const float*)d_in[21];
  const float* patterns = (const float*)d_in[22];
  const int*   epoch    = (const int*)d_in[23];
  const int*   totep    = (const int*)d_in[24];

  // Workspace (256B-aligned bump allocator)
  char* wp = (char*)d_ws;
  auto alloc = [&](size_t bytes) { char* r = wp; wp += (bytes + 255) & ~(size_t)255; return r; };
  float*  h       = (float*) alloc((size_t)BT * D * 4);        // fp32 residual stream
  __bf16* hnb     = (__bf16*)alloc((size_t)BT * D * 2);        // LN output (GEMM A)
  __bf16* xb      = (__bf16*)alloc((size_t)BT * IN * 2);       // x bf16 (later: h bf16)
  __bf16* qkvb    = (__bf16*)alloc((size_t)BT * 3 * D * 2);    // qkv bf16
  float*  scoresF = (float*) alloc((size_t)H * T * T * 4);     // scores (also hp/cos/qproj)
  __bf16* probsb  = (__bf16*)alloc((size_t)H * T * T * 2);     // probs bf16 (also lnq f32)
  __bf16* obufb   = (__bf16*)alloc((size_t)BT * D * 2);        // attn-out (also hpn/Q)
  __bf16* ffn1b   = (__bf16*)alloc((size_t)BT * 4 * D * 2);    // gelu out (also soft)
  __bf16* cbb     = (__bf16*)alloc((size_t)NS * D * 2);        // codebook bf16
  __bf16* cbnb    = (__bf16*)alloc((size_t)NS * D * 2);        // l2norm(codebook) bf16
  __bf16* pnb     = (__bf16*)alloc((size_t)NP * D * 2);        // l2norm(patterns) bf16
  __bf16* sym_wb  = (__bf16*)alloc((size_t)D * IN * 2);
  __bf16* ain_wb  = (__bf16*)alloc((size_t)L * 3 * D * D * 2);
  __bf16* aout_wb = (__bf16*)alloc((size_t)L * D * D * 2);
  __bf16* f_w1b   = (__bf16*)alloc((size_t)L * 4 * D * D * 2);
  __bf16* f_w2b   = (__bf16*)alloc((size_t)L * 4 * D * D * 2);
  __bf16* q_wb    = (__bf16*)alloc((size_t)D * D * 2);
  float*  scl     = (float*) alloc(256);

  // Output regions: emb | assign | logits | h | hard_idx
  float* out_emb    = (float*)d_out;
  float* out_assign = out_emb    + (long)BT * D;
  float* out_logits = out_assign + (long)BT * NP;
  float* out_h      = out_logits + (long)BT * NP;
  float* out_hard   = out_h      + (long)BT * D;

  // ---- One-time per launch: scale + bf16 conversions ---------------------
  vq_scale_kernel<<<1, 1, 0, stream>>>(epoch, totep, scl);
  cvt(stream, x,      xb,      (long)BT * IN);
  cvt(stream, sym_w,  sym_wb,  (long)D * IN);
  cvt(stream, codebook, cbb,   (long)NS * D);
  cvt(stream, ain_w,  ain_wb,  (long)L * 3 * D * D);
  cvt(stream, aout_w, aout_wb, (long)L * D * D);
  cvt(stream, f_w1,   f_w1b,   (long)L * 4 * D * D);
  cvt(stream, f_w2,   f_w2b,   (long)L * 4 * D * D);
  cvt(stream, q_w,    q_wb,    (long)D * D);
  l2norm_rows<__bf16><<<rowBlocks(NS), 256, 0, stream>>>(codebook, cbnb, NS, D);
  l2norm_rows<__bf16><<<rowBlocks(NP), 256, 0, stream>>>(patterns, pnb, NP, D);

  // ---- Symbolization -----------------------------------------------------
  float* hp = scoresF;   // hp/cos share the 32MB scores region (dead by layer 0)
  gemm(stream, xb, sym_wb, hp, false, sym_b, nullptr, nullptr, BT, D, IN,
       IN, 0,  1, IN, 0,  D, 0,  1, 1.0f, false);
  l2norm_rows<__bf16><<<rowBlocks(BT), 256, 0, stream>>>(hp, obufb, BT, D);   // hpn
  gemm(stream, obufb, cbnb, scoresF, false, nullptr, nullptr, nullptr, BT, NS, D,
       D, 0,  1, D, 0,  NS, 0,  1, 1.0f, false);
  softmax_rows<__bf16><<<rowBlocks(BT), 256, 0, stream>>>(scoresF, gumbel, ffn1b, BT, NS, 4.0f, 2.0f);
  gemm(stream, ffn1b, cbb, h, false, nullptr, nullptr, nullptr, BT, D, NS,
       NS, 0,  D, 1, 0,  D, 0,  1, 1.0f, false);
  {
    long tot = (long)BT * D;
    add_posenc<<<(int)((tot + 255) / 256), 256, 0, stream>>>(h, pos_enc, BT, T, D);
  }

  // ---- Transformer blocks ------------------------------------------------
  for (int i = 0; i < L; ++i) {
    const __bf16* inw  = ain_wb  + (long)i * 3 * D * D;
    const float*  inb  = ain_b   + (long)i * 3 * D;
    const __bf16* outw = aout_wb + (long)i * D * D;
    const float*  outb = aout_b  + (long)i * D;

    layernorm_rows<__bf16><<<rowBlocks(BT), 256, 0, stream>>>(h, n1_g + (long)i * D, n1_b + (long)i * D, hnb, BT, D);
    gemm(stream, hnb, inw, qkvb, true, inb, nullptr, nullptr, BT, 3 * D, D,
         D, 0,  1, D, 0,  3 * D, 0,  1, 1.0f, false);

    for (int b = 0; b < B; ++b) {
      const __bf16* qb = qkvb + (long)b * TQKV;           // q[b,t,h,:]
      const __bf16* kb = qkvb + (long)b * TQKV + D;       // k[b,s,h,:]
      const __bf16* vb = qkvb + (long)b * TQKV + 2 * D;   // v[b,s,h,:]
      // scores[h,t,s] = q.k  (batched over h) -> fp32 (staged path, K=64)
      gemm(stream, qb, kb, scoresF, false, nullptr, nullptr, nullptr, T, T, 64,
           3 * D, 64,  1, 3 * D, 64,  T, (long)T * T,  H, 1.0f, false);
      softmax_rows<__bf16><<<rowBlocks(H * T), 256, 0, stream>>>(scoresF, nullptr, probsb, H * T, T, 0.125f, 0.0f);
      // o[t,h,:] = probs @ v  (batched over h, N=64 -> direct kernel)
      gemm(stream, probsb, vb, obufb + (long)b * T * D, true, nullptr, nullptr, nullptr, T, 64, T,
           T, (long)T * T,  3 * D, 1, 64,  D, 64,  H, 1.0f, false);
    }
    gemm(stream, obufb, outw, h, false, outb, h, nullptr, BT, D, D,
         D, 0,  1, D, 0,  D, 0,  1, 1.0f, false);

    layernorm_rows<__bf16><<<rowBlocks(BT), 256, 0, stream>>>(h, n2_g + (long)i * D, n2_b + (long)i * D, hnb, BT, D);
    gemm(stream, hnb, f_w1b + (long)i * 4 * D * D, ffn1b, true, f_b1 + (long)i * 4 * D, nullptr, nullptr,
         BT, 4 * D, D,  D, 0,  1, D, 0,  4 * D, 0,  1, 1.0f, true);
    gemm(stream, ffn1b, f_w2b + (long)i * 4 * D * D, h, false, f_b2 + (long)i * D, h, nullptr,
         BT, D, 4 * D,  4 * D, 0,  1, 4 * D, 0,  D, 0,  1, 1.0f, false);
  }

  // ---- VQ head -----------------------------------------------------------
  __bf16* hb16 = xb;                 // reuse x region
  float*  qpF  = scoresF;            // reuse scores region
  float*  lnqF = (float*)probsb;     // reuse probs region (16MB, fits BT*D fp32)
  cvt(stream, h, hb16, (long)BT * D);
  gemm(stream, hb16, q_wb, qpF, false, q_b, nullptr, nullptr, BT, D, D,
       D, 0,  1, D, 0,  D, 0,  1, 1.0f, false);
  layernorm_rows<float><<<rowBlocks(BT), 256, 0, stream>>>(qpF, qln_g, qln_b, lnqF, BT, D);
  l2norm_rows<__bf16><<<rowBlocks(BT), 256, 0, stream>>>(lnqF, obufb, BT, D);   // Q
  gemm(stream, obufb, pnb, out_logits, false, nullptr, nullptr, scl, BT, NP, D,
       D, 0,  1, D, 0,  NP, 0,  1, 1.0f, false);
  vq_finalize<<<rowBlocks(BT), 256, 0, stream>>>(out_logits, patterns,
                                                 out_emb, out_assign, out_hard, BT, NP, D);
  hipMemcpyAsync(out_h, h, (size_t)BT * D * sizeof(float),
                 hipMemcpyDeviceToDevice, stream);
}
